// CombinedGCN_88167088652919
// MI455X (gfx1250) — compile-verified, matrix operand order checked by reading, not run on hardware
//
#include <hip/hip_runtime.h>
#include <hip/hip_bf16.h>

// ---------------------------------------------------------------------------
// CombinedGCN for MI455X (gfx1250, wave32, WMMA)
//
// Sizes (fixed by the reference): B=8, N_PER=50000 -> N=400000 nodes,
// F_IN=64, H1=128, EMB=64, E_total = B*E_PER = 400000 edges.
//
// Pipeline:
//   deg (atomics) -> dis=rsqrt(deg)
//   H1 = X @ W1            (WMMA f16->f32, fused AGG1 = H1*dis^2 self-loop init)
//   AGG1 += scatter(H1)    (hw float atomics, wave-per-edge, float4 lanes)
//   H2 = relu(AGG1+b1)@W2  (WMMA, bias+relu fused into A-tile staging,
//                           fused AGG2 = H2*dis^2)
//   AGG2 += scatter(H2)
//   pooled = mean_b relu(AGG2 + b2)
//   out = relu(pooled@fc_w+fc_b) @ out_w + out_b
//
// Roofline: GEMMs ~13 GFLOP (trivial under WMMA); scatter moves ~0.9 GB of
// gather+atomic traffic -> the run is bound by HBM/L2 atomic throughput, so
// scatters use hardware global_atomic_add_f32 (unsafeAtomicAdd: workspace is
// coarse-grained device memory) and fully coalesced float4 gathers.
//
// Workspace layout (floats):
//   deg/dis : N
//   H1      : N*128          (later reused: H2 = first N*64, AGG2 = next N*64)
//   AGG1    : N*128
//   pooled  : 512
// total ~ 411 MB.
// ---------------------------------------------------------------------------

typedef __attribute__((ext_vector_type(16))) _Float16 v16h;
typedef __attribute__((ext_vector_type(8)))  _Float16 v8h;
typedef __attribute__((ext_vector_type(4)))  _Float16 v4h;
typedef __attribute__((ext_vector_type(8)))  float    v8f;

#define N_NODES   400000
#define E_TOTAL   400000
#define N_PER_B   50000
#define ROWS_PER_BLOCK 128

// Hardware FP32 atomic add (global_atomic_add_f32). Workspace buffers are
// ordinary (coarse-grained) device memory, so the "unsafe" variant is safe
// here and avoids any CAS-loop fallback.
__device__ __forceinline__ void atomic_add_f32(float* p, float v) {
  unsafeAtomicAdd(p, v);
}

// ---------------------------------------------------------------- degree ----
__global__ __launch_bounds__(256) void deg_init_kernel(float* __restrict__ deg, int n) {
  int i = blockIdx.x * 256 + threadIdx.x;
  if (i < n) deg[i] = 1.0f;  // self-loop contribution
}

__global__ __launch_bounds__(256) void deg_count_kernel(const int* __restrict__ dst,
                                                        float* __restrict__ deg, int e) {
  int i = blockIdx.x * 256 + threadIdx.x;
  if (i < e) atomic_add_f32(&deg[dst[i]], 1.0f);
}

__global__ __launch_bounds__(256) void deg_rsqrt_kernel(float* __restrict__ deg, int n) {
  int i = blockIdx.x * 256 + threadIdx.x;
  if (i < n) deg[i] = rsqrtf(deg[i]);  // deg >= 1 always (self loop)
}

// ------------------------------------------------------------- WMMA GEMM ----
// H[row, n]   = sum_k A[row,k] * W[k,n]            (A optionally relu(A+bias))
// AGG[row, n] = H[row,n] * dis[row]^2              (self-loop init of aggregation)
//
// Block: 256 threads = 8 waves, 128 rows per block (16 rows per wave).
// W fragments packed once into LDS in the native 16x16x32 f16 B layout:
//   lane l (l16=l&15, hi=l>>4) holds column n0+l16,
//   half j -> k = ks*32 + (j>>3)*16 + hi*8 + (j&7).
// A fragments use the same k mapping with m = l16 (ISA 16-bit A 16x32 layout).
template <int K, int NOUT, bool IN_BIAS_RELU>
__global__ __launch_bounds__(256) void gemm_wmma_kernel(
    const float* __restrict__ X, const float* __restrict__ W,
    const float* __restrict__ bias_in, const float* __restrict__ dis,
    float* __restrict__ H, float* __restrict__ AGG) {
  constexpr int KSTEPS = K / 32;
  constexpr int NT     = NOUT / 16;

  __shared__ _Float16 ldsW[NOUT * K];              // fragment-packed
  __shared__ _Float16 ldsX[ROWS_PER_BLOCK * K];    // row-major f16 tile
  __shared__ float    ldsDis[ROWS_PER_BLOCK];

  const int tid  = threadIdx.x;
  const int row0 = blockIdx.x * ROWS_PER_BLOCK;

  // ---- pack W into B-fragment layout (W resident in L2; cheap) ----
  for (int idx = tid; idx < NOUT * K; idx += 256) {
    int j    = idx & 15;
    int lane = (idx >> 4) & 31;
    int frag = idx >> 9;               // 512 halves per fragment
    int ks   = frag % KSTEPS;
    int nt   = frag / KSTEPS;
    int hi   = lane >> 4, l16 = lane & 15;
    int k = ks * 32 + ((j >> 3) << 4) + hi * 8 + (j & 7);
    int n = nt * 16 + l16;
    ldsW[idx] = (_Float16)W[k * NOUT + n];
  }

  // ---- stage A tile f32 -> f16 (optionally fused bias+relu) ----
  for (int idx4 = tid; idx4 < (ROWS_PER_BLOCK * K) / 4; idx4 += 256) {
    int flat = idx4 * 4;
    int r = flat / K, k = flat % K;
    const float4 v = *(const float4*)(X + (size_t)(row0 + r) * K + k);
    float x0 = v.x, x1 = v.y, x2 = v.z, x3 = v.w;
    if constexpr (IN_BIAS_RELU) {
      x0 = fmaxf(x0 + bias_in[k + 0], 0.0f);
      x1 = fmaxf(x1 + bias_in[k + 1], 0.0f);
      x2 = fmaxf(x2 + bias_in[k + 2], 0.0f);
      x3 = fmaxf(x3 + bias_in[k + 3], 0.0f);
    }
    v4h h4 = {(_Float16)x0, (_Float16)x1, (_Float16)x2, (_Float16)x3};
    *(v4h*)(&ldsX[flat]) = h4;
  }
  if (tid < ROWS_PER_BLOCK) ldsDis[tid] = dis[row0 + tid];
  __syncthreads();

  const int wave = tid >> 5;
  const int lane = tid & 31;
  const int hi   = lane >> 4;
  const int l16  = lane & 15;

  // ---- load A fragments (two ds_load_b128 per fragment per lane) ----
  union Frag { v16h v; v8h h[2]; };
  Frag a[KSTEPS];
  const _Float16* arow = &ldsX[(wave * 16 + l16) * K + hi * 8];
#pragma unroll
  for (int ks = 0; ks < KSTEPS; ++ks) {
    a[ks].h[0] = *(const v8h*)(arow + ks * 32);
    a[ks].h[1] = *(const v8h*)(arow + ks * 32 + 16);
  }

#pragma unroll
  for (int nt = 0; nt < NT; ++nt) {
    v8f c = {};
#pragma unroll
    for (int ks = 0; ks < KSTEPS; ++ks) {
      v16h b = *(const v16h*)(&ldsW[((nt * KSTEPS + ks) * 32 + lane) * 16]);
      c = __builtin_amdgcn_wmma_f32_16x16x32_f16(
          /*neg_a=*/false, a[ks].v, /*neg_b=*/false, b,
          /*c_mod=*/(short)0, c, /*reuse_a=*/false, /*reuse_b=*/false);
    }
    // ---- epilogue: H and self-loop-initialized AGG ----
#pragma unroll
    for (int r = 0; r < 8; ++r) {
      int mrow = r + hi * 8;  // C/D layout: VGPR r, lanes16-31 -> M = r+8
      size_t grow = (size_t)(row0 + wave * 16 + mrow);
      float val = c[r];
      float ds  = ldsDis[wave * 16 + mrow];
      H[grow * NOUT + nt * 16 + l16]   = val;
      AGG[grow * NOUT + nt * 16 + l16] = val * ds * ds;
    }
  }
}

// --------------------------------------------------------- edge scatter -----
// One wave per edge; 32 lanes * VEC floats cover the C-wide feature row.
// Gather H[src] with a single b128/b64 load per lane; hardware float atomic
// add into AGG[dst] scaled by dis[s]*dis[d].
template <int C, int VEC>
__global__ __launch_bounds__(256) void scatter_kernel(
    const int* __restrict__ src, const int* __restrict__ dst,
    const float* __restrict__ dis, const float* __restrict__ H,
    float* __restrict__ AGG, int e_total) {
  int g    = blockIdx.x * 256 + threadIdx.x;
  int e    = g >> 5;
  int lane = g & 31;
  if (e >= e_total) return;
  int s = src[e];   // same address across the wave -> scalar-broadcast load
  int d = dst[e];
  float nrm = dis[s] * dis[d];
  const float* hp = H + (size_t)s * C + lane * VEC;
  float*       ap = AGG + (size_t)d * C + lane * VEC;
  if constexpr (VEC == 4) {
    const float4 v = *(const float4*)hp;          // global_load_b128
    atomic_add_f32(ap + 0, v.x * nrm);
    atomic_add_f32(ap + 1, v.y * nrm);
    atomic_add_f32(ap + 2, v.z * nrm);
    atomic_add_f32(ap + 3, v.w * nrm);
  } else {
    const float2 v = *(const float2*)hp;          // global_load_b64
    atomic_add_f32(ap + 0, v.x * nrm);
    atomic_add_f32(ap + 1, v.y * nrm);
  }
}

// ------------------------------------------------------------ mean pool -----
// grid = B blocks; block = (64 cols, 4 node-lanes). Fuses relu(AGG2 + b2).
__global__ __launch_bounds__(256) void pool_kernel(const float* __restrict__ AGG2,
                                                   const float* __restrict__ b2,
                                                   float* __restrict__ pooled) {
  const int b  = blockIdx.x;
  const int c  = threadIdx.x;   // 0..63
  const int ty = threadIdx.y;   // 0..3
  const float bc = b2[c];
  const size_t base = (size_t)b * N_PER_B * 64;
  float acc = 0.0f;
  for (int i = ty; i < N_PER_B; i += 4)
    acc += fmaxf(AGG2[base + (size_t)i * 64 + c] + bc, 0.0f);
  __shared__ float red[4][64];
  red[ty][c] = acc;
  __syncthreads();
  if (ty == 0) {
    float s = red[0][c] + red[1][c] + red[2][c] + red[3][c];
    pooled[b * 64 + c] = s * (1.0f / (float)N_PER_B);
  }
}

// ------------------------------------------------------------------ head ----
// One block, (64 cols, 8 batches): relu(pooled@fc_w+fc_b) @ out_w + out_b.
__global__ __launch_bounds__(512) void head_kernel(
    const float* __restrict__ pooled, const float* __restrict__ fc_w,
    const float* __restrict__ fc_b, const float* __restrict__ out_w,
    const float* __restrict__ out_b, float* __restrict__ out) {
  const int c = threadIdx.x;  // 0..63
  const int b = threadIdx.y;  // 0..7
  __shared__ float pl[8][64];
  __shared__ float hb[8][64];
  pl[b][c] = pooled[b * 64 + c];
  __syncthreads();
  float acc = fc_b[c];
#pragma unroll 8
  for (int k = 0; k < 64; ++k) acc += pl[b][k] * fc_w[k * 64 + c];
  hb[b][c] = fmaxf(acc, 0.0f);
  __syncthreads();
  float o = out_b[c];
#pragma unroll 8
  for (int k = 0; k < 64; ++k) o += hb[b][k] * out_w[k * 64 + c];
  out[b * 64 + c] = o;
}

// ---------------------------------------------------------------------------
extern "C" void kernel_launch(void* const* d_in, const int* in_sizes, int n_in,
                              void* d_out, int out_size, void* d_ws, size_t ws_size,
                              hipStream_t stream) {
  const float* X    = (const float*)d_in[0];  // [B, N_PER, 64]
  const int*   ei   = (const int*)  d_in[1];  // [B, 2, E_PER] flat int32
  const float* W1   = (const float*)d_in[2];  // [64, 128]
  const float* b1   = (const float*)d_in[3];  // [128]
  const float* W2   = (const float*)d_in[4];  // [128, 64]
  const float* b2   = (const float*)d_in[5];  // [64]
  const float* fc_w = (const float*)d_in[6];  // [64, 64]
  const float* fc_b = (const float*)d_in[7];  // [64]
  const float* ow   = (const float*)d_in[8];  // [64, 64]
  const float* ob   = (const float*)d_in[9];  // [64]
  float* out = (float*)d_out;                 // [8, 64]

  const int N = N_NODES, E = E_TOTAL;
  // reshape(2,-1) on contiguous (B,2,E_PER): src = first B*E_PER ints,
  // dst = second B*E_PER ints of the flat buffer.
  const int* srcI = ei;
  const int* dstI = ei + E;

  float* ws     = (float*)d_ws;
  float* deg    = ws;                            // N (deg -> dis in place)
  float* H1     = ws + N;                        // N*128
  float* AGG1   = H1 + (size_t)N * 128;          // N*128
  float* H2     = H1;                            // N*64 (H1 dead after scatter1)
  float* AGG2   = H1 + (size_t)N * 64;           // N*64
  float* pooled = AGG1 + (size_t)N * 128;        // 512

  deg_init_kernel<<<(N + 255) / 256, 256, 0, stream>>>(deg, N);
  deg_count_kernel<<<(E + 255) / 256, 256, 0, stream>>>(dstI, deg, E);
  deg_rsqrt_kernel<<<(N + 255) / 256, 256, 0, stream>>>(deg, N);

  // Layer 1: H1 = X @ W1 ; AGG1 = H1 * dis^2 (self loops)
  gemm_wmma_kernel<64, 128, false>
      <<<N / ROWS_PER_BLOCK, 256, 0, stream>>>(X, W1, nullptr, deg, H1, AGG1);
  scatter_kernel<128, 4>
      <<<(E * 32 + 255) / 256, 256, 0, stream>>>(srcI, dstI, deg, H1, AGG1, E);

  // Layer 2: H2 = relu(AGG1 + b1) @ W2 ; AGG2 = H2 * dis^2
  gemm_wmma_kernel<128, 64, true>
      <<<N / ROWS_PER_BLOCK, 256, 0, stream>>>(AGG1, W2, b1, deg, H2, AGG2);
  scatter_kernel<64, 2>
      <<<(E * 32 + 255) / 256, 256, 0, stream>>>(srcI, dstI, deg, H2, AGG2, E);

  pool_kernel<<<8, dim3(64, 4), 0, stream>>>(AGG2, b2, pooled);
  head_kernel<<<1, dim3(64, 8), 0, stream>>>(pooled, fc_w, fc_b, ow, ob, out);
}
